// NystromAttention_30923764531895
// MI455X (gfx1250) — compile-verified, matrix operand order checked by reading, not run on hardware
//
#include <hip/hip_runtime.h>
#include <math.h>

// ---------------- problem constants ----------------
#define B_   8
#define H_   8
#define S_   4096
#define D_   64
#define L_   64
#define KS_  33
#define BH_  (B_*H_)
#define SEG_ (S_/L_)            // 64
#define NSPLIT_ 8               // S-splits for the flash kv kernel
#define SCHUNK_ (S_/NSPLIT_)    // 512
#define SCALE_ 0.35355339059327373f   // 64^-0.25

// LDS matrix row stride in f16 elements (multiple of 8 -> 16B aligned rows)
#define LDSTR 72

typedef __attribute__((ext_vector_type(16))) _Float16 v16h;
typedef __attribute__((ext_vector_type(8)))  _Float16 v8h;
typedef __attribute__((ext_vector_type(8)))  float    v8f;

// ---------------- DPP16 half-wave (16-lane) reductions ----------------
// All row statistics live in a 16-lane half-wave (C-matrix layout), so the
// whole reduction is expressible in DPP16: xor1/xor2 via quad_perm, then
// half-row and full-row mirrors.  Pure VALU: no ds_bpermute, no DScnt waits.
template <int CTRL>
__device__ __forceinline__ float dpp_f(float v) {
  int r = __builtin_amdgcn_update_dpp(0, __float_as_int(v), CTRL, 0xF, 0xF, true);
  return __int_as_float(r);
}
__device__ __forceinline__ float red16_max(float v) {
  v = fmaxf(v, dpp_f<0xB1>(v));    // quad_perm [1,0,3,2]  (xor 1)
  v = fmaxf(v, dpp_f<0x4E>(v));    // quad_perm [2,3,0,1]  (xor 2)
  v = fmaxf(v, dpp_f<0x141>(v));   // row_half_mirror      (combine quads)
  v = fmaxf(v, dpp_f<0x140>(v));   // row_mirror           (combine octets)
  return v;
}
__device__ __forceinline__ float red16_sum(float v) {
  v += dpp_f<0xB1>(v);
  v += dpp_f<0x4E>(v);
  v += dpp_f<0x141>(v);
  v += dpp_f<0x140>(v);
  return v;
}

// ---------------- WMMA helpers (wave32, 16x16x32 f16 -> f32) ----------------
// A-fragment layout (ISA 7.12.2, 16-bit A 16x32):
//   lanes 0-15  : M=lane,    elems 0-7 = K 0..7,   elems 8-15 = K 16..23
//   lanes 16-31 : M=lane-16, elems 0-7 = K 8..15,  elems 8-15 = K 24..31
// B operands are stored transposed in LDS (row n = column n of B), so the
// same loader serves both A and B fragments.
__device__ __forceinline__ v16h load_frag(const _Float16* base, int r0, int kk) {
  int lane  = threadIdx.x & 31;
  int r     = r0 + (lane & 15);
  int khalf = lane >> 4;
  const _Float16* p = base + r * LDSTR + kk + khalf * 8;
  v8h lo = *(const v8h*)(p);
  v8h hi = *(const v8h*)(p + 16);
  v16h f;
#pragma unroll
  for (int i = 0; i < 8; ++i) { f[i] = lo[i]; f[i + 8] = hi[i]; }
  return f;
}

__device__ __forceinline__ v8f wmma16(v16h a, v16h b, v8f c) {
  return __builtin_amdgcn_wmma_f32_16x16x32_f16(false, a, false, b, (short)0, c,
                                                false, false);
}

__device__ __forceinline__ void zero_acc(v8f a[4]) {
#pragma unroll
  for (int nt = 0; nt < 4; ++nt)
#pragma unroll
    for (int j = 0; j < 8; ++j) a[nt][j] = 0.f;
}

// C = A * B^T (64x64).  One wave computes the 16x64 strip rows [m0,m0+16).
// All 10 fragments are hoisted before the 8 WMMAs so the ds_load_b128s
// pipeline against the matrix ops instead of serializing per tile.
__device__ __forceinline__ void mm_strip(const _Float16* lA, const _Float16* lB,
                                         int m0, v8f acc[4]) {
  v16h a0  = load_frag(lA, m0, 0);
  v16h a1  = load_frag(lA, m0, 32);
  v16h b00 = load_frag(lB,  0, 0);
  v16h b10 = load_frag(lB, 16, 0);
  v16h b20 = load_frag(lB, 32, 0);
  v16h b30 = load_frag(lB, 48, 0);
  v16h b01 = load_frag(lB,  0, 32);
  v16h b11 = load_frag(lB, 16, 32);
  v16h b21 = load_frag(lB, 32, 32);
  v16h b31 = load_frag(lB, 48, 32);
  acc[0] = wmma16(a0, b00, acc[0]);
  acc[1] = wmma16(a0, b10, acc[1]);
  acc[2] = wmma16(a0, b20, acc[2]);
  acc[3] = wmma16(a0, b30, acc[3]);
  acc[0] = wmma16(a1, b01, acc[0]);
  acc[1] = wmma16(a1, b11, acc[1]);
  acc[2] = wmma16(a1, b21, acc[2]);
  acc[3] = wmma16(a1, b31, acc[3]);
}

__device__ __forceinline__ void store_strip(_Float16* o, const v8f a[4],
                                            int m0, float s) {
  int lane = threadIdx.x & 31, hl = lane >> 4, n0 = lane & 15;
#pragma unroll
  for (int nt = 0; nt < 4; ++nt)
#pragma unroll
    for (int j = 0; j < 8; ++j)
      o[(m0 + hl * 8 + j) * LDSTR + nt * 16 + n0] = (_Float16)(a[nt][j] * s);
}
__device__ __forceinline__ void store_stripT(_Float16* o, const v8f a[4],
                                             int m0, float s) {
  int lane = threadIdx.x & 31, hl = lane >> 4, n0 = lane & 15;
#pragma unroll
  for (int nt = 0; nt < 4; ++nt)
#pragma unroll
    for (int j = 0; j < 8; ++j)
      o[(nt * 16 + n0) * LDSTR + m0 + hl * 8 + j] = (_Float16)(a[nt][j] * s);
}
// dst = (diag*I - C)^T  fused into the transposed store (Newton-Schulz steps)
__device__ __forceinline__ void store_stripT_diagsub(_Float16* o, const v8f a[4],
                                                     int m0, float diag) {
  int lane = threadIdx.x & 31, hl = lane >> 4, n0 = lane & 15;
#pragma unroll
  for (int nt = 0; nt < 4; ++nt) {
    int n = nt * 16 + n0;
#pragma unroll
    for (int j = 0; j < 8; ++j) {
      int m = m0 + hl * 8 + j;
      o[n * LDSTR + m] = (_Float16)(((m == n) ? diag : 0.f) - a[nt][j]);
    }
  }
}

// ---------------- kernel 0: init global-max scalar ----------------
__global__ void k_ginit(unsigned int* g) { *g = 0u; }

// ------------- kernel 1: landmarks (mask+scale fused means, b128 loads) ------
__global__ __launch_bounds__(64) void k_landmarks(const float* __restrict__ q,
                                                  const float* __restrict__ k,
                                                  const int* __restrict__ mask,
                                                  float* __restrict__ ql,
                                                  float* __restrict__ kl) {
  __shared__ float4 shq[64], shk[64];
  int bid = blockIdx.x;                 // BH_*L_
  int bh = bid >> 6, l = bid & 63;
  int b = bh / H_;
  int t = threadIdx.x;
  int cg = t & 15;                      // column group (4 cols)
  int g  = t >> 4;                      // row quarter
  const float4* qp = (const float4*)(q + ((size_t)bh * S_ + (size_t)l * SEG_) * D_);
  const float4* kp = (const float4*)(k + ((size_t)bh * S_ + (size_t)l * SEG_) * D_);
  const int*    mp = mask + (size_t)b * S_ + l * SEG_;
  float4 sq = {0.f, 0.f, 0.f, 0.f}, sk = {0.f, 0.f, 0.f, 0.f};
#pragma unroll 4
  for (int r = g * 16; r < g * 16 + 16; ++r) {
    float m = (float)mp[r];
    float4 qv = qp[r * 16 + cg];
    float4 kv = kp[r * 16 + cg];
    sq.x += qv.x * m; sq.y += qv.y * m; sq.z += qv.z * m; sq.w += qv.w * m;
    sk.x += kv.x * m; sk.y += kv.y * m; sk.z += kv.z * m; sk.w += kv.w * m;
  }
  shq[t] = sq; shk[t] = sk;
  __syncthreads();
  if (g == 0) {
    float c = SCALE_ / (float)SEG_;
    float4 aq = shq[t], ak = shk[t];
#pragma unroll
    for (int p = 1; p < 4; ++p) {
      float4 xq = shq[t + 16 * p], xk = shk[t + 16 * p];
      aq.x += xq.x; aq.y += xq.y; aq.z += xq.z; aq.w += xq.w;
      ak.x += xk.x; ak.y += xk.y; ak.z += xk.z; ak.w += xk.w;
    }
    aq.x *= c; aq.y *= c; aq.z *= c; aq.w *= c;
    ak.x *= c; ak.y *= c; ak.z *= c; ak.w *= c;
    ((float4*)(ql + ((size_t)bh * L_ + l) * D_))[cg] = aq;
    ((float4*)(kl + ((size_t)bh * L_ + l) * D_))[cg] = ak;
  }
}

// ---------------- kernel 2: K2 = softmax(ql·klT), colsum global max ----------
__global__ __launch_bounds__(128) void k_kernel2(const float* __restrict__ ql,
                                                 const float* __restrict__ kl,
                                                 float* __restrict__ k2out,
                                                 unsigned int* __restrict__ gmax) {
  __shared__ __align__(16) _Float16 lA[64 * LDSTR], lB[64 * LDSTR];
  __shared__ float lCS[64];
  int bh = blockIdx.x, tid = threadIdx.x;
  for (int i = tid; i < 1024; i += 128) {
    int m = i >> 4, c4 = (i & 15) * 4;
    float4 a = ((const float4*)(ql + (size_t)bh * 4096))[i];
    float4 bb = ((const float4*)(kl + (size_t)bh * 4096))[i];
    _Float16* pa = lA + m * LDSTR + c4;
    _Float16* pb = lB + m * LDSTR + c4;
    pa[0] = (_Float16)a.x;  pa[1] = (_Float16)a.y;
    pa[2] = (_Float16)a.z;  pa[3] = (_Float16)a.w;
    pb[0] = (_Float16)bb.x; pb[1] = (_Float16)bb.y;
    pb[2] = (_Float16)bb.z; pb[3] = (_Float16)bb.w;
  }
  if (tid < 64) lCS[tid] = 0.f;
  __syncthreads();

  int w = tid >> 5, lane = tid & 31, hl = lane >> 4;
  v8f acc[4]; zero_acc(acc);
  mm_strip(lA, lB, w * 16, acc);

#pragma unroll
  for (int j = 0; j < 8; ++j) {        // full row softmax over 64 landmarks
    float mx = -3.0e38f;
#pragma unroll
    for (int nt = 0; nt < 4; ++nt) mx = fmaxf(mx, acc[nt][j]);
    mx = red16_max(mx);
    float rs = 0.f;
#pragma unroll
    for (int nt = 0; nt < 4; ++nt) {
      float p = __expf(acc[nt][j] - mx);
      acc[nt][j] = p; rs += p;
    }
    rs = red16_sum(rs);
    float inv = 1.f / rs;
#pragma unroll
    for (int nt = 0; nt < 4; ++nt) acc[nt][j] *= inv;
  }
  float* op = k2out + (size_t)bh * 4096;
#pragma unroll
  for (int nt = 0; nt < 4; ++nt) {
    int n = nt * 16 + (lane & 15);
    float part = 0.f;
#pragma unroll
    for (int j = 0; j < 8; ++j) {
      int m = w * 16 + hl * 8 + j;
      op[m * 64 + n] = acc[nt][j];
      part += acc[nt][j];
    }
    part += __shfl_xor(part, 16, 32);  // combine the two half-waves
    if (hl == 0) atomicAdd(&lCS[n], part);   // ds_add_f32
  }
  __syncthreads();
  if (tid < 64)   // sums are positive -> uint-bit atomicMax == float max
    atomicMax(gmax, (unsigned int)__float_as_int(lCS[tid]));
}

// ---------------- kernel 3: Newton-Schulz pseudo-inverse (WMMA in LDS) -------
__global__ __launch_bounds__(128) void k_newton(const float* __restrict__ k2in,
                                                float* __restrict__ inv2,
                                                const unsigned int* __restrict__ gmax) {
  __shared__ __align__(16) _Float16 lK2[64 * LDSTR], lV[64 * LDSTR],
      lVt[64 * LDSTR], lKV[64 * LDSTR], lT[64 * LDSTR];
  int bh = blockIdx.x, tid = threadIdx.x, w = tid >> 5;
  float c = 1.f / __uint_as_float(*gmax);
  const float* kp = k2in + (size_t)bh * 4096;
  for (int i = tid; i < 4096; i += 128) {
    int m = i >> 6, n = i & 63;
    float v = kp[i];
    lK2[m * LDSTR + n] = (_Float16)v;        // K
    lVt[m * LDSTR + n] = (_Float16)(c * v);  // V^T = c*K
    lV[n * LDSTR + m]  = (_Float16)(c * v);  // V   = c*K^T
  }
  __syncthreads();

  v8f acc[4];
  for (int it = 0; it < 6; ++it) {
    zero_acc(acc); mm_strip(lK2, lVt, w * 16, acc);          // KV = K*V
    __syncthreads();
    store_strip(lKV, acc, w * 16, 1.f);
    store_stripT_diagsub(lT, acc, w * 16, 7.f);              // T = (7I-KV)^T
    __syncthreads();
    zero_acc(acc); mm_strip(lKV, lT, w * 16, acc);           // M1 = KV*(7I-KV)
    __syncthreads();
    store_stripT_diagsub(lT, acc, w * 16, 15.f);             // T = (15I-M1)^T
    __syncthreads();
    zero_acc(acc); mm_strip(lKV, lT, w * 16, acc);           // M3 = KV*M2
    __syncthreads();
    store_stripT_diagsub(lT, acc, w * 16, 13.f);             // T = (13I-M3)^T
    __syncthreads();
    zero_acc(acc); mm_strip(lV, lT, w * 16, acc);            // V*M4
    __syncthreads();
    store_strip (lV,  acc, w * 16, 0.25f);
    store_stripT(lVt, acc, w * 16, 0.25f);
    __syncthreads();
  }
  int lane = tid & 31, hl = lane >> 4;
  float* op = inv2 + (size_t)bh * 4096;
#pragma unroll
  for (int nt = 0; nt < 4; ++nt)
#pragma unroll
    for (int j = 0; j < 8; ++j)
      op[(w * 16 + hl * 8 + j) * 64 + nt * 16 + (lane & 15)] = acc[nt][j] * 0.25f;
}

// ------ kernel 4a: flash partials of softmax(ql·kT - mask)·v over S/8 -------
__global__ __launch_bounds__(128) void k_kv_part(const float* __restrict__ qlw,
                                                 const float* __restrict__ key,
                                                 const float* __restrict__ value,
                                                 const int* __restrict__ mask,
                                                 float* __restrict__ po,
                                                 float* __restrict__ pm,
                                                 float* __restrict__ pl) {
  __shared__ __align__(16) _Float16 lQ[64 * LDSTR], lK[64 * LDSTR],
      lVt[64 * LDSTR], lP[64 * LDSTR];
  __shared__ float lM[64];
  int blk = blockIdx.x;                       // BH_ * NSPLIT_
  int bh = blk >> 3, part = blk & (NSPLIT_ - 1);
  int b = bh / H_;
  int tid = threadIdx.x, w = tid >> 5, lane = tid & 31, hl = lane >> 4;

  for (int i = tid; i < 1024; i += 128) {
    int m = i >> 4, c4 = (i & 15) * 4;
    float4 a = ((const float4*)(qlw + (size_t)bh * 4096))[i];
    _Float16* p = lQ + m * LDSTR + c4;
    p[0] = (_Float16)a.x; p[1] = (_Float16)a.y;
    p[2] = (_Float16)a.z; p[3] = (_Float16)a.w;
  }
  v8f o[4]; zero_acc(o);
  float mrow[8], lrow[8];
#pragma unroll
  for (int j = 0; j < 8; ++j) { mrow[j] = -3.0e38f; lrow[j] = 0.f; }
  __syncthreads();

  for (int s0 = part * SCHUNK_; s0 < (part + 1) * SCHUNK_; s0 += 64) {
    for (int i = tid; i < 1024; i += 128) {
      int r = i >> 4, c4 = (i & 15) * 4, gs = s0 + r;
      float mk = (float)mask[(size_t)b * S_ + gs] * SCALE_;
      float4 kvv = ((const float4*)(key   + ((size_t)bh * S_ + gs) * D_))[i & 15];
      float4 vv  = ((const float4*)(value + ((size_t)bh * S_ + gs) * D_))[i & 15];
      _Float16* pk = lK + r * LDSTR + c4;
      pk[0] = (_Float16)(kvv.x * mk); pk[1] = (_Float16)(kvv.y * mk);
      pk[2] = (_Float16)(kvv.z * mk); pk[3] = (_Float16)(kvv.w * mk);
      lVt[(c4 + 0) * LDSTR + r] = (_Float16)vv.x;
      lVt[(c4 + 1) * LDSTR + r] = (_Float16)vv.y;
      lVt[(c4 + 2) * LDSTR + r] = (_Float16)vv.z;
      lVt[(c4 + 3) * LDSTR + r] = (_Float16)vv.w;
    }
    if (tid < 64) lM[tid] = (float)mask[(size_t)b * S_ + s0 + tid];
    __syncthreads();

    v8f sc[4]; zero_acc(sc);
    mm_strip(lQ, lK, w * 16, sc);               // scores = ql · k_chunk^T
#pragma unroll
    for (int nt = 0; nt < 4; ++nt) {
      float pen = -1.0e9f * (1.f - lM[nt * 16 + (lane & 15)]);
#pragma unroll
      for (int j = 0; j < 8; ++j) sc[nt][j] += pen;
    }
#pragma unroll
    for (int j = 0; j < 8; ++j) {               // online softmax update
      float mx = -3.0e38f;
#pragma unroll
      for (int nt = 0; nt < 4; ++nt) mx = fmaxf(mx, sc[nt][j]);
      mx = red16_max(mx);
      float nm = fmaxf(mrow[j], mx);
      float corr = __expf(mrow[j] - nm);
      float rs = 0.f;
#pragma unroll
      for (int nt = 0; nt < 4; ++nt) {
        float p = __expf(sc[nt][j] - nm);
        sc[nt][j] = p; rs += p;
      }
      rs = red16_sum(rs);
      lrow[j] = lrow[j] * corr + rs;
      mrow[j] = nm;
#pragma unroll
      for (int nt = 0; nt < 4; ++nt) o[nt][j] *= corr;
    }
    store_strip(lP, sc, w * 16, 1.f);           // own rows only
    mm_strip(lP, lVt, w * 16, o);               // O += P · V_chunk
    __syncthreads();                            // before chunk buffers reload
  }
  // write unnormalized partials (flash-decoding combine happens in k_kv_merge)
  float* pop = po + (size_t)(bh * NSPLIT_ + part) * 4096;
#pragma unroll
  for (int j = 0; j < 8; ++j) {
    int m = w * 16 + hl * 8 + j;
#pragma unroll
    for (int nt = 0; nt < 4; ++nt)
      pop[m * 64 + nt * 16 + (lane & 15)] = o[nt][j];
    if ((lane & 15) == 0) {
      pm[(size_t)(bh * NSPLIT_ + part) * 64 + m] = mrow[j];
      pl[(size_t)(bh * NSPLIT_ + part) * 64 + m] = lrow[j];
    }
  }
}

// ---------------- kernel 4b: merge flash partials -> kv ----------------
__global__ __launch_bounds__(128) void k_kv_merge(const float* __restrict__ po,
                                                  const float* __restrict__ pm,
                                                  const float* __restrict__ pl,
                                                  float* __restrict__ kv) {
  int bh = blockIdx.x, tid = threadIdx.x;
  for (int i = tid; i < 4096; i += 128) {
    int r = i >> 6, n = i & 63;
    float M = -3.0e38f;
#pragma unroll
    for (int p = 0; p < NSPLIT_; ++p)
      M = fmaxf(M, pm[(size_t)(bh * NSPLIT_ + p) * 64 + r]);
    float den = 0.f, num = 0.f;
#pragma unroll
    for (int p = 0; p < NSPLIT_; ++p) {
      float wgt = __expf(pm[(size_t)(bh * NSPLIT_ + p) * 64 + r] - M);
      den += wgt * pl[(size_t)(bh * NSPLIT_ + p) * 64 + r];
      num += wgt * po[(size_t)(bh * NSPLIT_ + p) * 4096 + r * 64 + n];
    }
    kv[(size_t)bh * 4096 + i] = num / den;
  }
}

// ---- kernel 5: X = softmax(q·klT)·inv2·kv + depthwise conv, fused output ----
__global__ __launch_bounds__(128) void k_main(const float* __restrict__ q,
                                              const float* __restrict__ value,
                                              const int* __restrict__ mask,
                                              const float* __restrict__ conv_w,
                                              const float* __restrict__ klw,
                                              const float* __restrict__ inv2,
                                              const float* __restrict__ kv,
                                              float* __restrict__ out) {
  __shared__ __align__(16) _Float16 lQ[64 * LDSTR], lKl[64 * LDSTR],
      lI2t[64 * LDSTR], lKvt[64 * LDSTR];
  __shared__ float lV96[96 * 65];   // rows s0-16 .. s0+79, masked v (conv slab)
  __shared__ float lCW[KS_ + 3];
  int bid = blockIdx.x;
  int bh = bid >> 6, scb = bid & 63, s0 = scb * 64;
  int b = bh / H_, h = bh % H_;
  int tid = threadIdx.x, w = tid >> 5, lane = tid & 31, hl = lane >> 4;

  for (int i = tid; i < 1024; i += 128) {
    int m = i >> 4, c4 = (i & 15) * 4;
    int gs = s0 + m;
    float mk = (float)mask[(size_t)b * S_ + gs] * SCALE_;
    float4 qv = ((const float4*)(q + ((size_t)bh * S_ + gs) * D_))[i & 15];
    float4 kl4 = ((const float4*)(klw  + (size_t)bh * 4096))[i];
    float4 i24 = ((const float4*)(inv2 + (size_t)bh * 4096))[i];
    float4 kv4 = ((const float4*)(kv   + (size_t)bh * 4096))[i];
    _Float16* pq = lQ  + m * LDSTR + c4;
    _Float16* pk = lKl + m * LDSTR + c4;
    pq[0] = (_Float16)(qv.x * mk); pq[1] = (_Float16)(qv.y * mk);
    pq[2] = (_Float16)(qv.z * mk); pq[3] = (_Float16)(qv.w * mk);
    pk[0] = (_Float16)kl4.x; pk[1] = (_Float16)kl4.y;
    pk[2] = (_Float16)kl4.z; pk[3] = (_Float16)kl4.w;
    lI2t[(c4 + 0) * LDSTR + m] = (_Float16)i24.x;   // inv2^T
    lI2t[(c4 + 1) * LDSTR + m] = (_Float16)i24.y;
    lI2t[(c4 + 2) * LDSTR + m] = (_Float16)i24.z;
    lI2t[(c4 + 3) * LDSTR + m] = (_Float16)i24.w;
    lKvt[(c4 + 0) * LDSTR + m] = (_Float16)kv4.x;   // kv^T
    lKvt[(c4 + 1) * LDSTR + m] = (_Float16)kv4.y;
    lKvt[(c4 + 2) * LDSTR + m] = (_Float16)kv4.z;
    lKvt[(c4 + 3) * LDSTR + m] = (_Float16)kv4.w;
  }
  for (int i = tid; i < 96 * 16; i += 128) {
    int r = i >> 4, c4 = (i & 15) * 4, gs = s0 - (KS_ / 2) + r;
    float4 v4 = {0.f, 0.f, 0.f, 0.f};
    if (gs >= 0 && gs < S_) {
      float mk = (float)mask[(size_t)b * S_ + gs];
      v4 = ((const float4*)(value + ((size_t)bh * S_ + gs) * D_))[i & 15];
      v4.x *= mk; v4.y *= mk; v4.z *= mk; v4.w *= mk;
    }
    float* pv = lV96 + r * 65 + c4;
    pv[0] = v4.x; pv[1] = v4.y; pv[2] = v4.z; pv[3] = v4.w;
  }
  if (tid < KS_) lCW[tid] = conv_w[h * KS_ + tid];
  __syncthreads();

  v8f acc[4]; zero_acc(acc);
  mm_strip(lQ, lKl, w * 16, acc);               // scores vs landmarks
  __syncthreads();                              // lKl about to be reused as P
#pragma unroll
  for (int j = 0; j < 8; ++j) {                 // softmax over 64 landmarks
    float mx = -3.0e38f;
#pragma unroll
    for (int nt = 0; nt < 4; ++nt) mx = fmaxf(mx, acc[nt][j]);
    mx = red16_max(mx);
    float rs = 0.f;
#pragma unroll
    for (int nt = 0; nt < 4; ++nt) {
      float p = __expf(acc[nt][j] - mx);
      acc[nt][j] = p; rs += p;
    }
    rs = red16_sum(rs);
    float inv = 1.f / rs;
#pragma unroll
    for (int nt = 0; nt < 4; ++nt) acc[nt][j] *= inv;
  }
  store_strip(lKl, acc, w * 16, 1.f);           // P1 (own rows; buffer reuse)
  zero_acc(acc);
  mm_strip(lKl, lI2t, w * 16, acc);             // W = P1 · inv2
  store_strip(lQ, acc, w * 16, 1.f);            // W  (own rows; buffer reuse)
  zero_acc(acc);
  mm_strip(lQ, lKvt, w * 16, acc);              // X = W · kv

  // fused depthwise conv: 40-row sliding register window per column,
  // taps in registers -> 160 LDS loads/lane instead of 1056.
  float cw[KS_];
#pragma unroll
  for (int t = 0; t < KS_; ++t) cw[t] = lCW[t];
  int mb = w * 16 + hl * 8;                     // first row of this 8-row group
#pragma unroll
  for (int nt = 0; nt < 4; ++nt) {
    int n = nt * 16 + (lane & 15);
    float win[40];
#pragma unroll
    for (int r = 0; r < 40; ++r) win[r] = lV96[(mb + r) * 65 + n];
#pragma unroll
    for (int j = 0; j < 8; ++j) {
      float cv = 0.f;
#pragma unroll
      for (int t = 0; t < KS_; ++t) cv = fmaf(cw[t], win[j + t], cv);
      out[((size_t)bh * S_ + s0 + mb + j) * D_ + n] = acc[nt][j] + cv;
    }
  }
}

// ---------------- host-side launch ----------------
extern "C" void kernel_launch(void* const* d_in, const int* in_sizes, int n_in,
                              void* d_out, int out_size, void* d_ws, size_t ws_size,
                              hipStream_t stream) {
  const float* q    = (const float*)d_in[0];
  const float* k    = (const float*)d_in[1];
  const float* v    = (const float*)d_in[2];
  const int*   mask = (const int*)d_in[3];
  const float* cw   = (const float*)d_in[4];
  float* out = (float*)d_out;
  float* ws  = (float*)d_ws;

  const size_t N64 = (size_t)BH_ * 64 * 64;   // 262144 floats per 64x64-per-bh
  float* ql   = ws;
  float* kl   = ws + N64;
  float* k2   = ws + 2 * N64;
  float* inv2 = ws + 3 * N64;
  float* kv   = ws + 4 * N64;
  unsigned int* gmax = (unsigned int*)(ws + 5 * N64);     // 64-float pad slot
  float* po = ws + 5 * N64 + 64;                          // BH*NSPLIT*64*64
  float* pm = po + (size_t)BH_ * NSPLIT_ * 4096;          // BH*NSPLIT*64
  float* pl = pm + (size_t)BH_ * NSPLIT_ * 64;            // BH*NSPLIT*64

  k_ginit    <<<1, 1, 0, stream>>>(gmax);
  k_landmarks<<<BH_ * L_, 64, 0, stream>>>(q, k, mask, ql, kl);
  k_kernel2  <<<BH_, 128, 0, stream>>>(ql, kl, k2, gmax);
  k_newton   <<<BH_, 128, 0, stream>>>(k2, inv2, gmax);
  k_kv_part  <<<BH_ * NSPLIT_, 128, 0, stream>>>(ql, k, v, mask, po, pm, pl);
  k_kv_merge <<<BH_, 128, 0, stream>>>(po, pm, pl, kv);
  k_main     <<<BH_ * (S_ / 64), 128, 0, stream>>>(q, v, mask, cw, kl, inv2, kv, out);
}
